// SwinTransformerBlock_72292889526798
// MI455X (gfx1250) — compile-verified, matrix operand order checked by reading, not run on hardware
//
#include <hip/hip_runtime.h>
#include <math.h>

typedef _Float16 half_t;
typedef __attribute__((ext_vector_type(16))) _Float16 v16h;
typedef __attribute__((ext_vector_type(8)))  _Float16 v8h;
typedef __attribute__((ext_vector_type(2)))  _Float16 v2h;
typedef __attribute__((ext_vector_type(8)))  float    v8f;

#define BB   2
#define LL   21952      // 28*28*28
#define CC   192
#define NWIN 128        // B * 4*4*4
#define NN   343        // 7^3
#define NP   352        // padded to 22*16
#define NHD  6
#define HDIM 32

// ---------------------------------------------------------------- helpers
__device__ __forceinline__ float gelu_f(float x) {
  return 0.5f * x * (1.0f + erff(x * 0.70710678118654752f));
}

__device__ __forceinline__ float wave_sum32(float v) {
#pragma unroll
  for (int m = 16; m >= 1; m >>= 1) v += __shfl_xor(v, m, 32);
  return v;
}
__device__ __forceinline__ float half_sum16(float v) {
#pragma unroll
  for (int m = 8; m >= 1; m >>= 1) v += __shfl_xor(v, m, 32);
  return v;
}
__device__ __forceinline__ float half_max16(float v) {
#pragma unroll
  for (int m = 8; m >= 1; m >>= 1) v = fmaxf(v, __shfl_xor(v, m, 32));
  return v;
}

__device__ __forceinline__ v16h combine_frag(v8h lo, v8h hv) {
  v16h f;
#pragma unroll
  for (int i = 0; i < 8; ++i) { f[i] = lo[i]; f[8 + i] = hv[i]; }
  return f;
}

// A/B fragment for V_WMMA_F32_16X16X32_F16 (ISA 7.12.2 16-bit layout):
// lane row = lane&15; lanes 0-15 hold K 0-7 / 16-23, lanes 16-31 hold
// K 8-15 / 24-31  ->  exactly two contiguous 16-byte runs per lane.
__device__ __forceinline__ v16h load_frag(const half_t* base, int ldk) {
  int lane = threadIdx.x & 31;
  int row  = lane & 15, hi = lane >> 4;
  const half_t* p = base + (size_t)row * ldk + hi * 8;
  v8h lo = *(const v8h*)p;
  v8h hv = *(const v8h*)(p + 16);
  return combine_frag(lo, hv);
}

#define WMMA_F16(A, B, C) \
  __builtin_amdgcn_wmma_f32_16x16x32_f16(false, (A), false, (B), (short)0, (C), false, false)

// async global->LDS 16B copy (CDNA5, tracked by ASYNCcnt)
__device__ __forceinline__ void async_b128(void* lds, const void* g) {
  unsigned loff = (unsigned)(unsigned long long)lds;
  asm volatile("global_load_async_to_lds_b128 %0, %1, off"
               :: "v"(loff), "v"(g) : "memory");
}
__device__ __forceinline__ void wait_async0() {
  asm volatile("s_wait_asynccnt 0x0" ::: "memory");
}

// ---------------------------------------------------------------- converts
__global__ void f32_to_f16_kernel(const float* __restrict__ in, half_t* __restrict__ out, int n) {
  for (int i = blockIdx.x * blockDim.x + threadIdx.x; i < n; i += gridDim.x * blockDim.x)
    out[i] = (half_t)in[i];
}

// w[O][Cin][27] -> out[tap][O][Cin]  (f16)
__global__ void conv_w_rearrange_kernel(const float* __restrict__ w, half_t* __restrict__ out,
                                        int O, int Cin) {
  int total = O * Cin * 27;
  for (int i = blockIdx.x * blockDim.x + threadIdx.x; i < total; i += gridDim.x * blockDim.x) {
    int o = i / (Cin * 27);
    int r = i % (Cin * 27);
    int c = r / 27, tap = r % 27;
    out[((size_t)tap * O + o) * Cin + c] = (half_t)w[i];
  }
}

// ---------------------------------------------------------------- layernorm
__global__ __launch_bounds__(256) void ln_kernel(const float* __restrict__ x,
                                                 const float* __restrict__ g,
                                                 const float* __restrict__ b,
                                                 half_t* __restrict__ out, int M) {
  int wid = threadIdx.x >> 5, lane = threadIdx.x & 31;
  int row = blockIdx.x * 8 + wid;
  if (row >= M) return;
  const float* xr = x + (size_t)row * CC;
  float v[6], s = 0.f;
#pragma unroll
  for (int i = 0; i < 6; ++i) { v[i] = xr[lane + 32 * i]; s += v[i]; }
  float mean = wave_sum32(s) * (1.f / CC);
  float var = 0.f;
#pragma unroll
  for (int i = 0; i < 6; ++i) { float d = v[i] - mean; var += d * d; }
  var = wave_sum32(var) * (1.f / CC);
  float rstd = rsqrtf(var + 1e-5f);
  half_t* orow = out + (size_t)row * CC;
#pragma unroll
  for (int i = 0; i < 6; ++i) {
    int c = lane + 32 * i;
    orow[c] = (half_t)((v[i] - mean) * rstd * g[c] + b[c]);
  }
}

// ---------------------------------------------------------------- WMMA GEMM
// O = A(M,K) @ W(N,K)^T + bias [; GELU][; + resid].  One wave = 16x64 tile
// (4 N-tiles share one A fragment).
template <bool GELU, bool OUTF16>
__global__ __launch_bounds__(256) void gemm_kernel(const half_t* __restrict__ A,
                                                   const half_t* __restrict__ W,
                                                   const float* __restrict__ bias,
                                                   const float* __restrict__ resid,
                                                   void* __restrict__ outp,
                                                   int M, int N, int K) {
  int wid = threadIdx.x >> 5, lane = threadIdx.x & 31;
  int mt = blockIdx.y * 8 + wid;
  int ng = blockIdx.x;                       // group of 4 N-tiles
  if (mt * 16 >= M) return;
  int col = lane & 15, hi = lane >> 4;
  const half_t* Abase = A + (size_t)(mt * 16) * K;
  const half_t* Wbase = W + (size_t)(ng * 64) * K;
  v8f acc[4] = {};
  for (int kb = 0; kb < K; kb += 32) {
    __builtin_prefetch(Abase + kb + 256, 0, 1);   // global_prefetch_b8
    v16h af = load_frag(Abase + kb, K);
#pragma unroll
    for (int j = 0; j < 4; ++j) {
      v16h bf = load_frag(Wbase + (size_t)(j * 16) * K + kb, K);
      acc[j] = WMMA_F16(af, bf, acc[j]);
    }
  }
#pragma unroll
  for (int j = 0; j < 4; ++j) {
    int n = ng * 64 + j * 16 + col;
    float bn = bias ? bias[n] : 0.f;
#pragma unroll
    for (int r = 0; r < 8; ++r) {
      int m = mt * 16 + r + hi * 8;
      float v = acc[j][r] + bn;
      if (GELU) v = gelu_f(v);
      if (resid) v += resid[(size_t)m * N + n];
      if (OUTF16) ((half_t*)outp)[(size_t)m * N + n] = (half_t)v;
      else        ((float*)outp)[(size_t)m * N + n] = v;
    }
  }
}

// ---------------------------------------------------------------- implicit-GEMM conv
// conv1: (B,L,192) f16 -> GELU -> (B,L,64) f16.  weights wt[tap][64][192].
// One wave = 16 positions x all 64 output channels.
__global__ __launch_bounds__(256) void conv1_kernel(const half_t* __restrict__ xn,
                                                    const half_t* __restrict__ wt,
                                                    const float* __restrict__ bias,
                                                    half_t* __restrict__ h1) {
  int wid = threadIdx.x >> 5, lane = threadIdx.x & 31;
  int col = lane & 15, hi = lane >> 4;
  int mt = blockIdx.y * 8 + wid;            // 0..3135
  int tIdx = mt & 1, bhw = mt >> 1;
  int b = bhw / 784, rem = bhw % 784;
  int h0 = rem / 28, w0 = rem % 28;
  int t0 = tIdx * 16;
  v8f acc[4] = {};
  v8h zz = {};
  for (int dh = -1; dh <= 1; ++dh) {
    int hh = h0 + dh;
    if ((unsigned)hh >= 28u) continue;       // wave-uniform
    for (int dw = -1; dw <= 1; ++dw) {
      int ww = w0 + dw;
      if ((unsigned)ww >= 28u) continue;     // wave-uniform
      for (int dt = -1; dt <= 1; ++dt) {
        int tap = (dh + 1) * 9 + (dw + 1) * 3 + (dt + 1);
        int tin = t0 + col + dt;             // per-lane M row
        bool tok = (unsigned)tin < 28u;
        int tc = tok ? tin : 0;              // clamp -> loads always legal
        const half_t* ab = xn + (size_t)((b * 784 + hh * 28 + ww) * 28 + tc) * CC + hi * 8;
        const half_t* wb = wt + (size_t)tap * 64 * CC;
        for (int cb = 0; cb < 6; ++cb) {
          v8h lo = *(const v8h*)(ab + cb * 32);
          v8h hv = *(const v8h*)(ab + cb * 32 + 16);
          if (!tok) { lo = zz; hv = zz; }    // branchless zero
          v16h af = combine_frag(lo, hv);
#pragma unroll
          for (int j = 0; j < 4; ++j) {
            v16h bf = load_frag(wb + (size_t)(j * 16) * CC + cb * 32, CC);
            acc[j] = WMMA_F16(af, bf, acc[j]);
          }
        }
      }
    }
  }
#pragma unroll
  for (int j = 0; j < 4; ++j) {
    int o = j * 16 + col;
    float bo = bias[o];
#pragma unroll
    for (int r = 0; r < 8; ++r) {
      int tout = t0 + r + hi * 8;
      if (tout < 28)
        h1[((size_t)(b * 784 + h0 * 28 + w0) * 28 + tout) * 64 + o] =
            (half_t)gelu_f(acc[j][r] + bo);
    }
  }
}

// conv2: (B,L,64) f16 -> (B,L,192) f32.  weights wt[tap][192][64].
__global__ __launch_bounds__(256) void conv2_kernel(const half_t* __restrict__ h1,
                                                    const half_t* __restrict__ wt,
                                                    const float* __restrict__ bias,
                                                    float* __restrict__ h2) {
  int wid = threadIdx.x >> 5, lane = threadIdx.x & 31;
  int col = lane & 15, hi = lane >> 4;
  int mt = blockIdx.y * 8 + wid;
  int og = blockIdx.x;                      // 0..2 (4 tiles each)
  int tIdx = mt & 1, bhw = mt >> 1;
  int b = bhw / 784, rem = bhw % 784;
  int h0 = rem / 28, w0 = rem % 28;
  int t0 = tIdx * 16;
  v8f acc[4] = {};
  v8h zz = {};
  for (int dh = -1; dh <= 1; ++dh) {
    int hh = h0 + dh;
    if ((unsigned)hh >= 28u) continue;
    for (int dw = -1; dw <= 1; ++dw) {
      int ww = w0 + dw;
      if ((unsigned)ww >= 28u) continue;
      for (int dt = -1; dt <= 1; ++dt) {
        int tap = (dh + 1) * 9 + (dw + 1) * 3 + (dt + 1);
        int tin = t0 + col + dt;
        bool tok = (unsigned)tin < 28u;
        int tc = tok ? tin : 0;
        const half_t* ab = h1 + (size_t)((b * 784 + hh * 28 + ww) * 28 + tc) * 64 + hi * 8;
        const half_t* wb = wt + ((size_t)tap * CC + og * 64) * 64;
        for (int cb = 0; cb < 2; ++cb) {
          v8h lo = *(const v8h*)(ab + cb * 32);
          v8h hv = *(const v8h*)(ab + cb * 32 + 16);
          if (!tok) { lo = zz; hv = zz; }
          v16h af = combine_frag(lo, hv);
#pragma unroll
          for (int j = 0; j < 4; ++j) {
            v16h bf = load_frag(wb + (size_t)(j * 16) * 64 + cb * 32, 64);
            acc[j] = WMMA_F16(af, bf, acc[j]);
          }
        }
      }
    }
  }
#pragma unroll
  for (int j = 0; j < 4; ++j) {
    int o = og * 64 + j * 16 + col;
    float bo = bias[o];
#pragma unroll
    for (int r = 0; r < 8; ++r) {
      int tout = t0 + r + hi * 8;
      if (tout < 28)
        h2[((size_t)(b * 784 + h0 * 28 + w0) * 28 + tout) * CC + o] = acc[j][r] + bo;
    }
  }
}

// ---------------------------------------------------------------- channel attention
__global__ __launch_bounds__(256) void ca_pool_kernel(const float* __restrict__ h2,
                                                      float* __restrict__ p) {
  int b = blockIdx.x / CC, c = blockIdx.x % CC;
  float s = 0.f;
  for (int l = threadIdx.x; l < LL; l += 256)
    s += h2[((size_t)b * LL + l) * CC + c];
  __shared__ float sm[8];
  s = wave_sum32(s);
  if ((threadIdx.x & 31) == 0) sm[threadIdx.x >> 5] = s;
  __syncthreads();
  if (threadIdx.x == 0) {
    float t = 0.f;
    for (int i = 0; i < 8; ++i) t += sm[i];
    p[b * CC + c] = t * (1.f / LL);
  }
}

__global__ void ca_mlp_kernel(const float* __restrict__ p,
                              const float* __restrict__ w1, const float* __restrict__ b1,
                              const float* __restrict__ w2, const float* __restrict__ b2,
                              float* __restrict__ a) {
  __shared__ float hid[BB][12];
  int tid = threadIdx.x;
  if (tid < BB * 12) {
    int b = tid / 12, j = tid % 12;
    float s = b1[j];
    for (int c = 0; c < CC; ++c) s += p[b * CC + c] * w1[j * CC + c];
    hid[b][j] = fmaxf(s, 0.f);
  }
  __syncthreads();
  if (tid < BB * CC) {
    int b = tid / CC, c = tid % CC;
    float s = b2[c];
    for (int j = 0; j < 12; ++j) s += hid[b][j] * w2[c * 12 + j];
    a[b * CC + c] = 1.f / (1.f + __expf(-s));
  }
}

// ---------------------------------------------------------------- attention
// one block per (window, head); 8 waves; dynamic LDS (~154 KB of 320 KB/WGP)
__global__ __launch_bounds__(256) void attn_kernel(const half_t* __restrict__ qkvx,
                                                   const half_t* __restrict__ qkvy,
                                                   const float* __restrict__ rpb,
                                                   half_t* __restrict__ att) {
  extern __shared__ char smem[];
  half_t* qs   = (half_t*)smem;            // [NP][32]
  half_t* ks   = qs + NP * 32;             // [NP][32]
  half_t* vT   = ks + NP * 32;             // [32][NP]
  half_t* Pall = vT + 32 * NP;             // [8][16][NP]

  int w = blockIdx.x, h = blockIdx.y;
  int b = w >> 6, wloc = w & 63;
  int hb = wloc >> 4, wb = (wloc >> 2) & 3, tb = wloc & 3;
  int tid = threadIdx.x;
  int wid = tid >> 5, lane = tid & 31;
  int col = lane & 15, hi = lane >> 4;
  const float scale = 0.17677669529663687f;   // 1/sqrt(32)

  // zero the padded rows/cols
  for (int idx = tid; idx < (NP - NN) * 32; idx += 256) {
    int n = NN + (idx >> 5), d = idx & 31;
    qs[n * 32 + d] = (half_t)0.f;
    ks[n * 32 + d] = (half_t)0.f;
  }
  for (int idx = tid; idx < 32 * (NP - NN); idx += 256) {
    int d = idx / (NP - NN), n = NN + idx % (NP - NN);
    vT[d * NP + n] = (half_t)0.f;
  }

  // async-copy q and k rows (pure 16B copies, ASYNCcnt tracked)
  for (int idx = tid; idx < NN * 4; idx += 256) {
    int n = idx >> 2, c = idx & 3;
    int i = n / 49, j = (n / 7) % 7, kk = n % 7;
    int l = ((hb * 7 + i) * 28 + (wb * 7 + j)) * 28 + (tb * 7 + kk);
    size_t r = (size_t)(b * LL + l) * 576;
    async_b128(qs + n * 32 + c * 8, qkvx + r + h * 32 + c * 8);
    async_b128(ks + n * 32 + c * 8, qkvy + r + 192 + h * 32 + c * 8);
  }
  // v^T (transposed scatter, plain stores)
  for (int idx = tid; idx < NN * 32; idx += 256) {
    int n = idx >> 5, d = idx & 31;
    int i = n / 49, j = (n / 7) % 7, kk = n % 7;
    int l = ((hb * 7 + i) * 28 + (wb * 7 + j)) * 28 + (tb * 7 + kk);
    size_t r = (size_t)(b * LL + l) * 576;
    vT[d * NP + n] = qkvy[r + 384 + h * 32 + d];
  }
  wait_async0();
  __syncthreads();

  for (int mt = wid; mt < 22; mt += 8) {
    half_t* P = Pall + (size_t)wid * 16 * NP;
    v16h qa = load_frag(qs + mt * 16 * 32, 32);

    // S = scale * q k^T + rpb  (f16 rows in LDS)
    for (int nt = 0; nt < 22; ++nt) {
      v16h kb = load_frag(ks + nt * 16 * 32, 32);
      v8f c = {};
      c = WMMA_F16(qa, kb, c);
#pragma unroll
      for (int r = 0; r < 8; ++r) {
        int ml = r + hi * 8;
        int m = mt * 16 + ml;
        int n = nt * 16 + col;
        float s = c[r] * scale;
        if (m < NN && n < NN) {
          int iq = m / 49, jq = (m / 7) % 7, kq = m % 7;
          int ik = n / 49, jk = (n / 7) % 7, kn = n % 7;
          int ridx = ((iq - ik + 6) * 13 + (jq - jk + 6)) * 13 + (kq - kn + 6);
          s += rpb[ridx * NHD + h];
        } else {
          s = -1.0e30f;
        }
        P[ml * NP + n] = (half_t)s;
      }
    }

    // softmax per row (vectorized LDS passes)
    float rinv[8];
#pragma unroll
    for (int r = 0; r < 8; ++r) {
      int ml = r + hi * 8;
      half_t* Prow = P + ml * NP;
      float mx = -3.0e38f;
      for (int t = col; t < NP / 2; t += 16) {
        v2h e2 = *(const v2h*)(Prow + 2 * t);
        mx = fmaxf(mx, fmaxf((float)e2[0], (float)e2[1]));
      }
      mx = half_max16(mx);
      float sum = 0.f;
      for (int t = col; t < NP / 2; t += 16) {
        v2h e2 = *(v2h*)(Prow + 2 * t);
        float e0 = __expf((float)e2[0] - mx);
        float e1 = __expf((float)e2[1] - mx);
        e2[0] = (half_t)e0; e2[1] = (half_t)e1;
        *(v2h*)(Prow + 2 * t) = e2;
        sum += e0 + e1;
      }
      rinv[r] = 1.f / half_sum16(sum);
    }

    // O = P @ V  (two 16-wide d tiles), normalize, store
    for (int dt = 0; dt < 2; ++dt) {
      v8f o = {};
      for (int kt = 0; kt < 11; ++kt) {
        v16h pa = load_frag(P + kt * 32, NP);
        v16h vb = load_frag(vT + (size_t)(dt * 16) * NP + kt * 32, NP);
        o = WMMA_F16(pa, vb, o);
      }
#pragma unroll
      for (int r = 0; r < 8; ++r) {
        int m = mt * 16 + r + hi * 8;
        if (m < NN)
          att[((size_t)w * NN + m) * CC + h * 32 + dt * 16 + col] =
              (half_t)(o[r] * rinv[r]);
      }
    }
  }
}

// ---------------------------------------------------------------- residual + window reverse
__global__ void resid_kernel(const float* __restrict__ x, const float* __restrict__ attp,
                             const float* __restrict__ h2, const float* __restrict__ a,
                             float* __restrict__ xres) {
  size_t total = (size_t)BB * LL * CC;
  for (size_t i = blockIdx.x * (size_t)blockDim.x + threadIdx.x; i < total;
       i += (size_t)gridDim.x * blockDim.x) {
    int c = (int)(i % CC);
    int l = (int)((i / CC) % LL);
    int b = (int)(i / ((size_t)CC * LL));
    int hh = l / 784, ww = (l / 28) % 28, tt = l % 28;
    int wi = ((b * 4 + hh / 7) * 4 + ww / 7) * 4 + tt / 7;
    int n = ((hh % 7) * 7 + (ww % 7)) * 7 + (tt % 7);
    xres[i] = x[i] + attp[((size_t)wi * NN + n) * CC + c] + h2[i] * a[b * CC + c];
  }
}

// ---------------------------------------------------------------- launch
extern "C" void kernel_launch(void* const* d_in, const int* in_sizes, int n_in,
                              void* d_out, int out_size, void* d_ws, size_t ws_size,
                              hipStream_t stream) {
  (void)in_sizes; (void)n_in; (void)out_size; (void)ws_size;
  const float* x      = (const float*)d_in[0];
  const float* y      = (const float*)d_in[1];
  const float* n1g    = (const float*)d_in[3];
  const float* n1b    = (const float*)d_in[4];
  const float* qkv_w  = (const float*)d_in[5];
  const float* qkv_b  = (const float*)d_in[6];
  const float* proj_w = (const float*)d_in[7];
  const float* proj_b = (const float*)d_in[8];
  const float* rpb    = (const float*)d_in[9];
  const float* n2g    = (const float*)d_in[10];
  const float* n2b    = (const float*)d_in[11];
  const float* fc1_w  = (const float*)d_in[12];
  const float* fc1_b  = (const float*)d_in[13];
  const float* fc2_w  = (const float*)d_in[14];
  const float* fc2_b  = (const float*)d_in[15];
  const float* cab1_w = (const float*)d_in[16];
  const float* cab1_b = (const float*)d_in[17];
  const float* cab2_w = (const float*)d_in[18];
  const float* cab2_b = (const float*)d_in[19];
  const float* ca1_w  = (const float*)d_in[20];
  const float* ca1_b  = (const float*)d_in[21];
  const float* ca2_w  = (const float*)d_in[22];
  const float* ca2_b  = (const float*)d_in[23];

  const int M = BB * LL;                       // 43904 rows

  char* wp = (char*)d_ws;
  auto alloc = [&](size_t bytes) -> char* {
    char* p = wp;
    wp += (bytes + 255) & ~(size_t)255;
    return p;
  };
  half_t* qkv_wh = (half_t*)alloc((size_t)576 * CC * 2);
  half_t* proj_wh= (half_t*)alloc((size_t)CC * CC * 2);
  half_t* fc1_wh = (half_t*)alloc((size_t)768 * CC * 2);
  half_t* fc2_wh = (half_t*)alloc((size_t)CC * 768 * 2);
  half_t* w1h    = (half_t*)alloc((size_t)27 * 64 * CC * 2);
  half_t* w2h    = (half_t*)alloc((size_t)27 * CC * 64 * 2);
  half_t* xn     = (half_t*)alloc((size_t)M * CC * 2);
  half_t* yn     = (half_t*)alloc((size_t)M * CC * 2);
  half_t* qkvx   = (half_t*)alloc((size_t)M * 576 * 2);
  half_t* qkvy   = (half_t*)alloc((size_t)M * 576 * 2);
  half_t* h1     = (half_t*)alloc((size_t)M * 64 * 2);
  float*  h2     = (float*)alloc((size_t)M * CC * 4);
  float*  pbuf   = (float*)alloc((size_t)BB * CC * 4);
  float*  abuf   = (float*)alloc((size_t)BB * CC * 4);
  half_t* att    = (half_t*)alloc((size_t)M * CC * 2);
  float*  attp   = (float*)alloc((size_t)M * CC * 4);
  float*  xres   = (float*)alloc((size_t)M * CC * 4);
  half_t* hln2   = (half_t*)alloc((size_t)M * CC * 2);
  half_t* ff1    = (half_t*)alloc((size_t)M * 768 * 2);

  // 1) weight conversion
  auto cvt = [&](const float* src, half_t* dst, int n) {
    int g = (n + 255) / 256; if (g > 4096) g = 4096;
    f32_to_f16_kernel<<<g, 256, 0, stream>>>(src, dst, n);
  };
  cvt(qkv_w, qkv_wh, 576 * CC);
  cvt(proj_w, proj_wh, CC * CC);
  cvt(fc1_w, fc1_wh, 768 * CC);
  cvt(fc2_w, fc2_wh, CC * 768);
  conv_w_rearrange_kernel<<<1296, 256, 0, stream>>>(cab1_w, w1h, 64, CC);
  conv_w_rearrange_kernel<<<1296, 256, 0, stream>>>(cab2_w, w2h, CC, 64);

  // 2) layernorm 1
  ln_kernel<<<(M + 7) / 8, 256, 0, stream>>>(x, n1g, n1b, xn, M);
  ln_kernel<<<(M + 7) / 8, 256, 0, stream>>>(y, n1g, n1b, yn, M);

  // 3) CAB conv branch (implicit-GEMM WMMA)
  conv1_kernel<<<dim3(1, 392), 256, 0, stream>>>(xn, w1h, cab1_b, h1);
  conv2_kernel<<<dim3(3, 392), 256, 0, stream>>>(h1, w2h, cab2_b, h2);
  ca_pool_kernel<<<BB * CC, 256, 0, stream>>>(h2, pbuf);
  ca_mlp_kernel<<<1, 384, 0, stream>>>(pbuf, ca1_w, ca1_b, ca2_w, ca2_b, abuf);

  // 4) QKV projections
  gemm_kernel<false, true><<<dim3(9, 343), 256, 0, stream>>>(xn, qkv_wh, qkv_b, nullptr,
                                                             qkvx, M, 576, CC);
  gemm_kernel<false, true><<<dim3(9, 343), 256, 0, stream>>>(yn, qkv_wh, qkv_b, nullptr,
                                                             qkvy, M, 576, CC);

  // 5) windowed attention
  size_t att_lds = (size_t)(3 * NP * 32 + 8 * 16 * NP) * sizeof(half_t);  // ~154 KB
  attn_kernel<<<dim3(NWIN, NHD), 256, att_lds, stream>>>(qkvx, qkvy, rpb, att);

  // 6) output projection (windowed rows)
  gemm_kernel<false, false><<<dim3(3, 343), 256, 0, stream>>>(att, proj_wh, proj_b, nullptr,
                                                              attp, M, CC, CC);

  // 7) residual combine (+ window reverse + CA scale)
  resid_kernel<<<4096, 256, 0, stream>>>(x, attp, h2, abuf, xres);

  // 8) MLP
  ln_kernel<<<(M + 7) / 8, 256, 0, stream>>>(xres, n2g, n2b, hln2, M);
  gemm_kernel<true, true><<<dim3(12, 343), 256, 0, stream>>>(hln2, fc1_wh, fc1_b, nullptr,
                                                             ff1, M, 768, CC);
  gemm_kernel<false, false><<<dim3(3, 343), 256, 0, stream>>>(ff1, fc2_wh, fc2_b, xres,
                                                              (float*)d_out, M, CC, 768);
}
